// NodeViewReadoutFFN_forEvenRank_87634512707843
// MI455X (gfx1250) — matrix operand order repaired and checked by reading, not compiled, not run on hardware
//
#include <hip/hip_runtime.h>

typedef unsigned short ushort_t;
typedef __attribute__((ext_vector_type(16))) __bf16 v16bf;
typedef __attribute__((ext_vector_type(8)))  float  v8f;

#define N_ATOMS 200000
#define N_NBR   10
#define N_MOLS  8000
#define HID     128
#define FDIM    151
#define FEAT    200
#define DFF     512
#define FFN_H   256
#define TASKS   12
#define APM     25      // atoms per molecule (200000/8000)

#define K1      288     // padded x width: 151 feat + 1 pad + 128 aggr + 8 pad
#define A_STR   296     // LDS row stride (mult of 8 -> 16B aligned rows)
#define H_STR   520
#define K2      352     // padded mol row: 128 + 200 + 24 pad
#define A2_STR  360
#define H2_STR  264

__device__ __forceinline__ ushort_t f2bf(float f) {
    union { float f; unsigned u; } x; x.f = f;
    unsigned r = x.u + 0x7FFFu + ((x.u >> 16) & 1u);
    return (ushort_t)(r >> 16);
}

union Frag { uint4 u[2]; v16bf v; };

// ---------------------------------------------------------------------------
// Kernel 0: convert weights to bf16 in WMMA-tiled layout, zero mol accumulator.
// Tiled layout for B operands: tile (kt,nt) holds 32 lanes x 16 elems; lane l, elem s
// -> n = nt*16 + l%16, k = kt*32 + 16*(s/8) + 8*(l/16) + s%8 (lane-contiguous 32B read).
// ---------------------------------------------------------------------------
__global__ void k0_prep(const float* __restrict__ W1, const float* __restrict__ W2,
                        const float* __restrict__ Wf1, const float* __restrict__ Wf2,
                        ushort_t* __restrict__ W1b, ushort_t* __restrict__ W2b,
                        ushort_t* __restrict__ Wf1b, ushort_t* __restrict__ Wf2b,
                        float* __restrict__ molacc)
{
    const int stride = gridDim.x * blockDim.x;
    const int tid = blockIdx.x * blockDim.x + threadIdx.x;

    // W1b: K1 x DFF, NT = 32. x layout: [0..150]=feat, 151=0, [152..279]=aggr, rest 0
    for (int i = tid; i < K1 * DFF; i += stride) {
        int s = i & 15, l = (i >> 4) & 31, t = i >> 9;
        int nr = l & 15, kr = ((s & 8) << 1) + ((l >> 4) << 3) + (s & 7);
        int nt = t & 31, kt = t >> 5;
        int k = kt * 32 + kr, n = nt * 16 + nr;
        float v = 0.f;
        if (k < FDIM) v = W1[k * DFF + n];
        else if (k > FDIM && k < FDIM + 1 + HID) v = W1[(k - 1) * DFF + n];
        W1b[i] = f2bf(v);
    }
    // W2b: DFF x HID, NT = 8
    for (int i = tid; i < DFF * HID; i += stride) {
        int s = i & 15, l = (i >> 4) & 31, t = i >> 9;
        int nr = l & 15, kr = ((s & 8) << 1) + ((l >> 4) << 3) + (s & 7);
        int nt = t & 7, kt = t >> 3;
        int k = kt * 32 + kr, n = nt * 16 + nr;
        W2b[i] = f2bf(W2[k * HID + n]);
    }
    // Wf1b: K2 x FFN_H, NT = 16
    for (int i = tid; i < K2 * FFN_H; i += stride) {
        int s = i & 15, l = (i >> 4) & 31, t = i >> 9;
        int nr = l & 15, kr = ((s & 8) << 1) + ((l >> 4) << 3) + (s & 7);
        int nt = t & 15, kt = t >> 4;
        int k = kt * 32 + kr, n = nt * 16 + nr;
        float v = (k < HID + FEAT) ? Wf1[k * FFN_H + n] : 0.f;
        Wf1b[i] = f2bf(v);
    }
    // Wf2b: FFN_H x 16 (NT = 1, same tiled layout), cols 12..15 zero
    for (int i = tid; i < FFN_H * 16; i += stride) {
        int s = i & 15, l = (i >> 4) & 31;
        int k = (i >> 9) * 32 + ((s & 8) << 1) + ((l >> 4) << 3) + (s & 7), n = l & 15;
        float v = (n < TASKS) ? Wf2[k * TASKS + n] : 0.f;
        Wf2b[i] = f2bf(v);
    }
    for (int i = tid; i < N_MOLS * HID; i += stride) molacc[i] = 0.f;
}

// ---------------------------------------------------------------------------
// Kernel 1: fused gather-sum + GEMM1(relu) + GEMM2 + layernorm + segment atomics
// 256 threads (8 waves), M-tile = 128 atoms, wave w owns rows 16w..16w+15.
// ---------------------------------------------------------------------------
__global__ __launch_bounds__(256) void k1_atoms(
    const float* __restrict__ atom_out, const float* __restrict__ orig,
    const int* __restrict__ a2a,
    const ushort_t* __restrict__ W1b, const float* __restrict__ b1,
    const ushort_t* __restrict__ W2b, const float* __restrict__ b2,
    const float* __restrict__ gma, const float* __restrict__ bta,
    float* __restrict__ molacc)
{
    __shared__ ushort_t sA[128 * A_STR];   // 75,776 B
    __shared__ ushort_t sH[128 * H_STR];   // 133,120 B

    const int wave = threadIdx.x >> 5;
    const int lane = threadIdx.x & 31;
    const int half = lane >> 4;
    const int nc   = lane & 15;
    const int tile0 = blockIdx.x * 128;

    // ---- Phase 1: gather + concat -> bf16 A tile ----
    for (int rr = 0; rr < 16; ++rr) {
        const int r = rr * 8 + wave;
        const int atom = tile0 + r;
        ushort_t* arow = &sA[r * A_STR];
        if (atom < N_ATOMS) {
            #pragma unroll
            for (int it = 0; it < 5; ++it) {
                int cc = it * 32 + lane;
                if (cc < FDIM) arow[cc] = f2bf(orig[atom * FDIM + cc]);
            }
            float4 acc = make_float4(0.f, 0.f, 0.f, 0.f);
            const int* nb = &a2a[atom * N_NBR];
            #pragma unroll
            for (int j = 0; j < N_NBR; ++j) {
                const int nidx = nb[j];   // wave-uniform -> s_load
                const float4 vv = *(const float4*)&atom_out[nidx * HID + lane * 4];
                acc.x += vv.x; acc.y += vv.y; acc.z += vv.z; acc.w += vv.w;
            }
            ushort_t* ap = &arow[152 + lane * 4];  // 8B-aligned
            ap[0] = f2bf(acc.x); ap[1] = f2bf(acc.y);
            ap[2] = f2bf(acc.z); ap[3] = f2bf(acc.w);
            if (lane == 0) arow[FDIM] = 0;          // pad col 151
            if (lane < 8)  arow[280 + lane] = 0;    // pad cols 280..287
        } else {
            for (int cc = lane; cc < K1; cc += 32) arow[cc] = 0;
        }
    }
    __syncthreads();

    // ---- Phase 2: h = relu(x @ W1 + b1) -> bf16 in sH ----
    Frag afr[9];
    {
        const ushort_t* abase = &sA[(16 * wave + nc) * A_STR + half * 8];
        #pragma unroll
        for (int kt = 0; kt < 9; ++kt) {
            afr[kt].u[0] = *(const uint4*)(abase + kt * 32);
            afr[kt].u[1] = *(const uint4*)(abase + kt * 32 + 16);
        }
    }
    for (int nt = 0; nt < 32; ++nt) {
        v8f acc = {0.f,0.f,0.f,0.f,0.f,0.f,0.f,0.f};
        #pragma unroll
        for (int kt = 0; kt < 9; ++kt) {
            Frag b;
            const ushort_t* bp = &W1b[((kt * 32 + nt) * 32 + lane) * 16];
            b.u[0] = *(const uint4*)(bp);
            b.u[1] = *(const uint4*)(bp + 8);
            acc = __builtin_amdgcn_wmma_f32_16x16x32_bf16(false, afr[kt].v, false, b.v,
                                                          (short)0, acc, false, false);
        }
        const float bias = b1[nt * 16 + nc];
        #pragma unroll
        for (int v = 0; v < 8; ++v) {
            float hv = acc[v] + bias;
            hv = fmaxf(hv, 0.f);
            const int row = 16 * wave + v + 8 * half;
            sH[row * H_STR + nt * 16 + nc] = f2bf(hv);
        }
    }
    __syncthreads();

    // ---- Phase 3: c = h @ W2 ----
    const v8f vzero = {0.f,0.f,0.f,0.f,0.f,0.f,0.f,0.f};
    v8f c[8];
    #pragma unroll
    for (int nt = 0; nt < 8; ++nt) c[nt] = vzero;
    for (int kt = 0; kt < 16; ++kt) {
        Frag a2;
        const ushort_t* ap = &sH[(16 * wave + nc) * H_STR + kt * 32 + half * 8];
        a2.u[0] = *(const uint4*)(ap);
        a2.u[1] = *(const uint4*)(ap + 16);
        #pragma unroll
        for (int nt = 0; nt < 8; ++nt) {
            Frag b;
            const ushort_t* bp = &W2b[((kt * 8 + nt) * 32 + lane) * 16];
            b.u[0] = *(const uint4*)(bp);
            b.u[1] = *(const uint4*)(bp + 8);
            c[nt] = __builtin_amdgcn_wmma_f32_16x16x32_bf16(false, a2.v, false, b.v,
                                                            (short)0, c[nt], false, false);
        }
    }

    // ---- Phase 4: +b2, layernorm (shfl within 16-lane half), segment atomics ----
    float b2v[8], gv[8], bv[8];
    #pragma unroll
    for (int nt = 0; nt < 8; ++nt) {
        b2v[nt] = b2[nt * 16 + nc];
        gv[nt]  = gma[nt * 16 + nc];
        bv[nt]  = bta[nt * 16 + nc];
    }
    const int atom_base = tile0 + 16 * wave + 8 * half;
    int curmol = -1;
    float psum[8] = {0.f,0.f,0.f,0.f,0.f,0.f,0.f,0.f};
    #pragma unroll
    for (int v = 0; v < 8; ++v) {
        float vals[8];
        float s1 = 0.f, s2 = 0.f;
        #pragma unroll
        for (int nt = 0; nt < 8; ++nt) {
            const float x = c[nt][v] + b2v[nt];
            vals[nt] = x; s1 += x; s2 += x * x;
        }
        #pragma unroll
        for (int off = 1; off <= 8; off <<= 1) {
            s1 += __shfl_xor(s1, off, 32);
            s2 += __shfl_xor(s2, off, 32);
        }
        const float mu = s1 * (1.f / (float)HID);
        const float var = s2 * (1.f / (float)HID) - mu * mu;
        const float rstd = rsqrtf(var + 1e-6f);
        const int atom = atom_base + v;
        const int mol = (atom < N_ATOMS) ? (atom / APM) : -1;
        if (mol != curmol) {
            if (curmol >= 0) {
                #pragma unroll
                for (int nt = 0; nt < 8; ++nt) {
                    atomicAdd(&molacc[curmol * HID + nt * 16 + nc], psum[nt]);
                    psum[nt] = 0.f;
                }
            }
            curmol = mol;
        }
        if (mol >= 0) {
            #pragma unroll
            for (int nt = 0; nt < 8; ++nt)
                psum[nt] += gv[nt] * (vals[nt] - mu) * rstd + bv[nt];
        }
    }
    if (curmol >= 0) {
        #pragma unroll
        for (int nt = 0; nt < 8; ++nt)
            atomicAdd(&molacc[curmol * HID + nt * 16 + nc], psum[nt]);
    }
}

// ---------------------------------------------------------------------------
// Kernel 2: molecule FFN: out = relu([mol/size ‖ feat] @ Wf1 + bf1) @ Wf2 + bf2
// ---------------------------------------------------------------------------
__global__ __launch_bounds__(256) void k2_mols(
    const float* __restrict__ molacc, const int* __restrict__ a_scope,
    const float* __restrict__ feats,
    const ushort_t* __restrict__ Wf1b, const float* __restrict__ bf1,
    const ushort_t* __restrict__ Wf2b, const float* __restrict__ bf2,
    float* __restrict__ out)
{
    __shared__ ushort_t sA[128 * A2_STR];  // 92,160 B
    __shared__ ushort_t sH[128 * H2_STR];  // 67,584 B

    const int wave = threadIdx.x >> 5;
    const int lane = threadIdx.x & 31;
    const int half = lane >> 4;
    const int nc   = lane & 15;
    const int tile0 = blockIdx.x * 128;

    // ---- Phase 1: build bf16 A tile ----
    for (int rr = 0; rr < 16; ++rr) {
        const int r = rr * 8 + wave;
        const int m = tile0 + r;
        ushort_t* arow = &sA[r * A2_STR];
        if (m < N_MOLS) {
            const float inv = 1.f / (float)a_scope[m * 2 + 1];
            const float4 vv = *(const float4*)&molacc[m * HID + lane * 4];
            ushort_t* ap = &arow[lane * 4];
            ap[0] = f2bf(vv.x * inv); ap[1] = f2bf(vv.y * inv);
            ap[2] = f2bf(vv.z * inv); ap[3] = f2bf(vv.w * inv);
            #pragma unroll
            for (int it = 0; it < 7; ++it) {
                int cc = it * 32 + lane;
                if (cc < FEAT) arow[HID + cc] = f2bf(feats[m * FEAT + cc]);
            }
            if (lane < 24) arow[HID + FEAT + lane] = 0;  // pad 328..351
        } else {
            for (int cc = lane; cc < K2; cc += 32) arow[cc] = 0;
        }
    }
    __syncthreads();

    // ---- Phase 2: h = relu(A @ Wf1 + bf1) ----
    Frag afr[11];
    {
        const ushort_t* abase = &sA[(16 * wave + nc) * A2_STR + half * 8];
        #pragma unroll
        for (int kt = 0; kt < 11; ++kt) {
            afr[kt].u[0] = *(const uint4*)(abase + kt * 32);
            afr[kt].u[1] = *(const uint4*)(abase + kt * 32 + 16);
        }
    }
    for (int nt = 0; nt < 16; ++nt) {
        v8f acc = {0.f,0.f,0.f,0.f,0.f,0.f,0.f,0.f};
        #pragma unroll
        for (int kt = 0; kt < 11; ++kt) {
            Frag b;
            const ushort_t* bp = &Wf1b[((kt * 16 + nt) * 32 + lane) * 16];
            b.u[0] = *(const uint4*)(bp);
            b.u[1] = *(const uint4*)(bp + 8);
            acc = __builtin_amdgcn_wmma_f32_16x16x32_bf16(false, afr[kt].v, false, b.v,
                                                          (short)0, acc, false, false);
        }
        const float bias = bf1[nt * 16 + nc];
        #pragma unroll
        for (int v = 0; v < 8; ++v) {
            float hv = acc[v] + bias;
            hv = fmaxf(hv, 0.f);
            const int row = 16 * wave + v + 8 * half;
            sH[row * H2_STR + nt * 16 + nc] = f2bf(hv);
        }
    }
    __syncthreads();

    // ---- Phase 3: out = h @ Wf2 + bf2 (N padded to 16; pad cols unstored) ----
    v8f cacc = {0.f,0.f,0.f,0.f,0.f,0.f,0.f,0.f};
    #pragma unroll
    for (int kt = 0; kt < 8; ++kt) {
        Frag a2;
        const ushort_t* ap = &sH[(16 * wave + nc) * H2_STR + kt * 32 + half * 8];
        a2.u[0] = *(const uint4*)(ap);
        a2.u[1] = *(const uint4*)(ap + 16);
        Frag b;
        const ushort_t* bp = &Wf2b[(kt * 32 + lane) * 16];
        b.u[0] = *(const uint4*)(bp);
        b.u[1] = *(const uint4*)(bp + 8);
        cacc = __builtin_amdgcn_wmma_f32_16x16x32_bf16(false, a2.v, false, b.v,
                                                       (short)0, cacc, false, false);
    }
    const float bias = (nc < TASKS) ? bf2[nc] : 0.f;
    #pragma unroll
    for (int v = 0; v < 8; ++v) {
        const int m = tile0 + 16 * wave + v + 8 * half;
        if (m < N_MOLS && nc < TASKS) out[m * TASKS + nc] = cacc[v] + bias;
    }
}

// ---------------------------------------------------------------------------
extern "C" void kernel_launch(void* const* d_in, const int* in_sizes, int n_in,
                              void* d_out, int out_size, void* d_ws, size_t ws_size,
                              hipStream_t stream)
{
    (void)in_sizes; (void)n_in; (void)out_size; (void)ws_size;
    const float* atom_out = (const float*)d_in[0];
    const float* orig     = (const float*)d_in[1];
    const float* feats    = (const float*)d_in[2];
    const int*   a2a      = (const int*)d_in[3];
    const int*   a_scope  = (const int*)d_in[4];
    const float* W1  = (const float*)d_in[5];
    const float* b1  = (const float*)d_in[6];
    const float* W2  = (const float*)d_in[7];
    const float* b2  = (const float*)d_in[8];
    const float* gma = (const float*)d_in[9];
    const float* bta = (const float*)d_in[10];
    const float* Wf1 = (const float*)d_in[11];
    const float* bf1 = (const float*)d_in[12];
    const float* Wf2 = (const float*)d_in[13];
    const float* bf2 = (const float*)d_in[14];
    float* out = (float*)d_out;

    char* ws = (char*)d_ws;
    ushort_t* W1b  = (ushort_t*)(ws);                               // 294,912 B
    ushort_t* W2b  = (ushort_t*)(ws + 294912);                      // 131,072 B
    ushort_t* Wf1b = (ushort_t*)(ws + 294912 + 131072);             // 180,224 B
    ushort_t* Wf2b = (ushort_t*)(ws + 294912 + 131072 + 180224);    //   8,192 B
    float* molacc  = (float*)(ws + 294912 + 131072 + 180224 + 8192);// 4,096,000 B

    k0_prep<<<512, 256, 0, stream>>>(W1, W2, Wf1, Wf2, W1b, W2b, Wf1b, Wf2b, molacc);
    k1_atoms<<<(N_ATOMS + 127) / 128, 256, 0, stream>>>(
        atom_out, orig, a2a, W1b, b1, W2b, b2, gma, bta, molacc);
    k2_mols<<<(N_MOLS + 127) / 128, 256, 0, stream>>>(
        molacc, a_scope, feats, Wf1b, bf1, Wf2b, bf2, out);
}
